// AblatedGATTransformer_55353538511019
// MI455X (gfx1250) — matrix operand order repaired and checked by reading, not compiled, not run on hardware
//
#include <hip/hip_runtime.h>
#include <hip/hip_bf16.h>
#include <math.h>

// ---- problem constants -----------------------------------------------------
#define Bc   512
#define Sc   72
#define Nc   5
#define Fc   10
#define Hc   4
#define Cc   32
#define Dc   128
#define Lc   3
#define FFc  256
#define Kc   6
#define NSc  5
#define Gc   (Bc*Sc)          // 36864 tokens / graphs
#define EPSc 1e-5f

typedef _Float16 half_t;
typedef _Float16 v16h __attribute__((ext_vector_type(16)));
typedef _Float16 v8h  __attribute__((ext_vector_type(8)));
typedef float    v8f  __attribute__((ext_vector_type(8)));

// ---- wave32 helpers --------------------------------------------------------
static __device__ __forceinline__ float wave_sum32(float v) {
#pragma unroll
  for (int m = 16; m > 0; m >>= 1) v += __shfl_xor(v, m, 32);
  return v;
}

// ---- WMMA fragment loaders (per CDNA5 ISA 7.12.2 layouts) ------------------
// A: 16x32 f16, row-major source with leading dim ldk (elements).
// lanes 0-15: row M=lane, elems {k0..k0+7, k0+16..k0+23}
// lanes 16-31: row M=lane-16, elems {k0+8..k0+15, k0+24..k0+31}
static __device__ __forceinline__ v16h load_a_frag(const half_t* base, int ldk,
                                                   int lane, int k0) {
  const int r  = lane & 15;
  const int hi = lane >> 4;
  const int ka = k0 + hi * 8;
  const int kb = ka + 16;
  v8h a0 = *(const v8h*)(base + (size_t)r * ldk + ka);
  v8h a1 = *(const v8h*)(base + (size_t)r * ldk + kb);
  v16h f;
#pragma unroll
  for (int i = 0; i < 8; ++i) { f[i] = a0[i]; f[i + 8] = a1[i]; }
  return f;
}

// B: 32x16 f16 where B[k][n] = W[n][k], W row-major [N,K] with leading dim ldk.
// lanes 0-15: column N=lane, K = {k0..k0+7, k0+16..k0+23} ; lanes 16-31: column N=lane-16, K = {k0+8..k0+15, k0+24..k0+31}
static __device__ __forceinline__ v16h load_b_frag(const half_t* wbase, int ldk,
                                                   int lane, int k0) {
  const int c  = lane & 15;
  const int hi = lane >> 4;
  const int kc = k0 + hi * 8;
  v8h b0 = *(const v8h*)(wbase + (size_t)c * ldk + kc);
  v8h b1 = *(const v8h*)(wbase + (size_t)c * ldk + kc + 16);
  v16h f;
#pragma unroll
  for (int i = 0; i < 8; ++i) { f[i] = b0[i]; f[i + 8] = b1[i]; }
  return f;
}

// ---- CDNA5 async global->LDS staging of the shared A tile ------------------
// Copies 16 rows x K halfs (16-byte granules) with GLOBAL_LOAD_ASYNC_TO_LDS
// (ASYNCcnt-tracked DMA, no VGPR round trip). Caller must __syncthreads()
// after this returns (this wave's transfers are already drained here).
static __device__ __forceinline__ void stage_a_async(const half_t* abase,
                                                     half_t* sA, int K,
                                                     int tid) {
  const int chunks_per_row = K / 8;          // 16B chunks per row
  const int chunks = 16 * chunks_per_row;
  for (int c = tid; c < chunks; c += 256) {
    const int row = c / chunks_per_row;
    const int col = (c % chunks_per_row) * 8;
    unsigned long long ga =
        (unsigned long long)(uintptr_t)(abase + (size_t)row * K + col);
    unsigned int la = (unsigned int)(uintptr_t)(sA + row * K + col);
    asm volatile("global_load_async_to_lds_b128 %0, %1, off"
                 :: "v"(la), "v"(ga) : "memory");
  }
  asm volatile("s_wait_asynccnt 0" ::: "memory");
}

// ---- kernel 0: f32 -> f16 weight mirror ------------------------------------
__global__ void __launch_bounds__(256) f32_to_f16_kernel(
    const float* __restrict__ src, half_t* __restrict__ dst, int n) {
  int i = blockIdx.x * 256 + threadIdx.x;
  if (i < n) dst[i] = (half_t)src[i];
}

// ---- kernel 1: fused GAT + node-mean + positional encoding -----------------
// One wave per graph (8 graphs per 256-thread block). Graph is dense K5 + self
// loops, so attention for each dst is a softmax over all 5 sources.
__global__ void __launch_bounds__(256) gat_pe_kernel(
    const float* __restrict__ x, const float* __restrict__ gat_w,
    const float* __restrict__ att_src, const float* __restrict__ att_dst,
    const float* __restrict__ gat_bias, const float* __restrict__ ln_g,
    const float* __restrict__ ln_b,
    float* __restrict__ t32, half_t* __restrict__ t16) {
  __shared__ float s_w[Fc][Dc];
  __shared__ float s_as[Dc], s_ad[Dc], s_bias[Dc], s_g[Dc], s_b[Dc];
  __shared__ float s_x[8][Nc][Fc];

  const int tid = threadIdx.x, lane = tid & 31, wv = tid >> 5;
  for (int i = tid; i < Fc * Dc; i += 256) s_w[i / Dc][i % Dc] = gat_w[i];
  for (int i = tid; i < Dc; i += 256) {
    s_as[i] = att_src[i]; s_ad[i] = att_dst[i];
    s_bias[i] = gat_bias[i]; s_g[i] = ln_g[i]; s_b[i] = ln_b[i];
  }
  const int g = blockIdx.x * 8 + wv;
  for (int i = lane; i < Nc * Fc; i += 32)
    s_x[wv][i / Fc][i % Fc] = x[(size_t)g * Nc * Fc + i];
  __syncthreads();

  // h[n][j]: lane owns columns d_j = lane + 32*j  (head j, channel = lane)
  float h[Nc][4];
#pragma unroll
  for (int n = 0; n < Nc; ++n)
#pragma unroll
    for (int j = 0; j < 4; ++j) {
      float acc = 0.f;
#pragma unroll
      for (int f = 0; f < Fc; ++f) acc += s_x[wv][n][f] * s_w[f][lane + 32 * j];
      h[n][j] = acc;
    }

  float asrc[Nc][4], adst[Nc][4];
#pragma unroll
  for (int n = 0; n < Nc; ++n)
#pragma unroll
    for (int j = 0; j < 4; ++j) {
      asrc[n][j] = wave_sum32(h[n][j] * s_as[lane + 32 * j]);
      adst[n][j] = wave_sum32(h[n][j] * s_ad[lane + 32 * j]);
    }

  float tacc[4] = {0.f, 0.f, 0.f, 0.f};
#pragma unroll
  for (int dn = 0; dn < Nc; ++dn) {
    float vals[4];
    float lsum = 0.f;
#pragma unroll
    for (int j = 0; j < 4; ++j) {
      float e[Nc], mx = -3.0e38f;
#pragma unroll
      for (int s = 0; s < Nc; ++s) {
        float v = asrc[s][j] + adst[dn][j];
        e[s] = v > 0.f ? v : 0.2f * v;       // leaky_relu(0.2)
        mx = fmaxf(mx, e[s]);
      }
      float den = 0.f, ag = 0.f;
#pragma unroll
      for (int s = 0; s < Nc; ++s) {
        float ex = expf(e[s] - mx);
        den += ex;
        ag  += ex * h[s][j];
      }
      vals[j] = ag / den + s_bias[lane + 32 * j];
      lsum += vals[j];
    }
    // LayerNorm over D=128 (4 vals per lane across 32 lanes)
    float mean = wave_sum32(lsum) * (1.f / Dc);
    float vs = 0.f;
#pragma unroll
    for (int j = 0; j < 4; ++j) { float d = vals[j] - mean; vs += d * d; }
    float rstd = rsqrtf(wave_sum32(vs) * (1.f / Dc) + EPSc);
#pragma unroll
    for (int j = 0; j < 4; ++j) {
      int d = lane + 32 * j;
      tacc[j] += ((vals[j] - mean) * rstd * s_g[d] + s_b[d]) * (1.f / Nc);
    }
  }

  // positional encoding, write f32 + f16 mirrors
  const int pos = g % Sc;
#pragma unroll
  for (int j = 0; j < 4; ++j) {
    int d = lane + 32 * j;
    float div = expf(-logf(10000.f) * (float)(d & ~1) / (float)Dc);
    float ang = (float)pos * div;
    float pe  = (d & 1) ? cosf(ang) : sinf(ang);
    float tv  = tacc[j] + pe;
    t32[(size_t)g * Dc + d] = tv;
    t16[(size_t)g * Dc + d] = (half_t)tv;
  }
}

// ---- kernel 2: generic WMMA GEMM, out = [relu](A @ W^T + bias), f16 out ----
// A [M,K] f16 row-major, W [N,K] f16 row-major. Block: 16 rows x 128 cols
// (8 waves, one 16x16 tile each). A tile staged once to LDS via async DMA;
// B streams from L2 as global_load_b128. grid = (M/16, N/128)
__global__ void __launch_bounds__(256) gemm_bias_f16_kernel(
    const half_t* __restrict__ A, const half_t* __restrict__ W,
    const float* __restrict__ bias, half_t* __restrict__ out,
    int M, int K, int N, int do_relu) {
  __shared__ half_t sA[16 * 256];            // up to K=256
  const int tid = threadIdx.x;
  const int lane = tid & 31, wv = tid >> 5;
  const int m0 = blockIdx.x * 16;
  const int n0 = blockIdx.y * 128 + wv * 16;

  stage_a_async(A + (size_t)m0 * K, sA, K, tid);

  const float bv = bias[n0 + (lane & 15)];
  v8f acc;
#pragma unroll
  for (int r = 0; r < 8; ++r) acc[r] = bv;
  const half_t* wbase = W + (size_t)n0 * K;
  __syncthreads();                            // LDS A tile visible to all waves
  for (int k0 = 0; k0 < K; k0 += 32) {
    v16h af = load_a_frag(sA, K, lane, k0);   // ds_load_b128 pairs
    v16h bf = load_b_frag(wbase, K, lane, k0);
    acc = __builtin_amdgcn_wmma_f32_16x16x32_f16(false, af, false, bf,
                                                 (short)0, acc, false, false);
  }
  const int col = n0 + (lane & 15);
  const int rb  = m0 + ((lane >> 4) * 8);
#pragma unroll
  for (int r = 0; r < 8; ++r) {
    float v = acc[r];
    if (do_relu) v = v > 0.f ? v : 0.f;
    out[(size_t)(rb + r) * N + col] = (half_t)v;
  }
}

// ---- kernel 3: WMMA GEMM fused with residual + LayerNorm -------------------
// t = LN(t + A @ W^T + bias); N fixed to 128. Updates t32/t16 in place.
__global__ void __launch_bounds__(256) gemm_res_ln_kernel(
    const half_t* __restrict__ A, const half_t* __restrict__ W,
    const float* __restrict__ bias, const float* __restrict__ gamma,
    const float* __restrict__ beta,
    float* __restrict__ t32, half_t* __restrict__ t16, int K) {
  __shared__ half_t sA[16 * 256];
  __shared__ float tile[16][Dc];
  const int tid = threadIdx.x;
  const int lane = tid & 31, wv = tid >> 5;
  const int m0 = blockIdx.x * 16;
  const int n0 = wv * 16;

  // pull residual rows toward the caches while the WMMA loop runs
  __builtin_prefetch(t32 + (size_t)(m0 + (tid >> 4)) * Dc + (tid & 15) * 8,
                     0, 3);

  stage_a_async(A + (size_t)m0 * K, sA, K, tid);

  const float bv = bias[n0 + (lane & 15)];
  v8f acc;
#pragma unroll
  for (int r = 0; r < 8; ++r) acc[r] = bv;
  const half_t* wbase = W + (size_t)n0 * K;
  __syncthreads();                            // LDS A tile visible to all waves
  for (int k0 = 0; k0 < K; k0 += 32) {
    v16h af = load_a_frag(sA, K, lane, k0);
    v16h bf = load_b_frag(wbase, K, lane, k0);
    acc = __builtin_amdgcn_wmma_f32_16x16x32_f16(false, af, false, bf,
                                                 (short)0, acc, false, false);
  }
  const int col = n0 + (lane & 15);
  const int rb  = (lane >> 4) * 8;
#pragma unroll
  for (int r = 0; r < 8; ++r) tile[rb + r][col] = acc[r];
  __syncthreads();

  // LayerNorm epilogue: wave wv owns rows {2wv, 2wv+1}
#pragma unroll
  for (int rr = wv * 2; rr < wv * 2 + 2; ++rr) {
    const int grow = m0 + rr;
    float v[4], s = 0.f;
#pragma unroll
    for (int j = 0; j < 4; ++j) {
      int d = lane + 32 * j;
      float xv = tile[rr][d] + t32[(size_t)grow * Dc + d];  // residual
      v[j] = xv; s += xv;
    }
    float mean = wave_sum32(s) * (1.f / Dc);
    float vs = 0.f;
#pragma unroll
    for (int j = 0; j < 4; ++j) { float d = v[j] - mean; vs += d * d; }
    float rstd = rsqrtf(wave_sum32(vs) * (1.f / Dc) + EPSc);
#pragma unroll
    for (int j = 0; j < 4; ++j) {
      int d = lane + 32 * j;
      float y = (v[j] - mean) * rstd * gamma[d] + beta[d];
      t32[(size_t)grow * Dc + d] = y;
      t16[(size_t)grow * Dc + d] = (half_t)y;
    }
  }
}

// ---- kernel 4: attention for one (head, batch) -----------------------------
// qkv [G, 384] f16. Stages q,k (80x32, zero-padded) and v transposed (32x96)
// in LDS; WMMA scores; f32 softmax; WMMA P.V; writes O f16 [G,128].
__global__ void __launch_bounds__(256) attn_kernel(
    const half_t* __restrict__ qkv, half_t* __restrict__ o) {
  __shared__ half_t qs[80][Cc];
  __shared__ half_t ks_[80][Cc];
  __shared__ half_t vt[Cc][96];     // vt[c][s] = V[s][c]
  __shared__ float  sc[80][80];
  __shared__ half_t pl[80][96];     // softmax probs, zero-padded
  const int h = blockIdx.x, b = blockIdx.y;
  const int tid = threadIdx.x, lane = tid & 31, wv = tid >> 5;

  for (int i = tid; i < 80 * Cc; i += 256) {
    int row = i >> 5, c = i & 31;
    half_t qv = (half_t)0.f, kv = (half_t)0.f;
    if (row < Sc) {
      size_t base = ((size_t)b * Sc + row) * (3 * Dc) + h * Cc + c;
      qv = qkv[base];
      kv = qkv[base + Dc];
    }
    qs[row][c] = qv; ks_[row][c] = kv;
  }
  for (int i = tid; i < Cc * 96; i += 256) {
    int c = i / 96, s = i % 96;
    vt[c][s] = (s < Sc)
        ? qkv[((size_t)b * Sc + s) * (3 * Dc) + 2 * Dc + h * Cc + c]
        : (half_t)0.f;
  }
  __syncthreads();

  const float scale = 0.17677669529663687f;   // 1/sqrt(32)
  for (int t = wv; t < 25; t += 8) {          // 5x5 tiles of 80x80 scores
    int mt = t / 5, nt = t % 5;
    v8f acc;
#pragma unroll
    for (int r = 0; r < 8; ++r) acc[r] = 0.f;
    v16h af = load_a_frag(&qs[mt * 16][0], Cc, lane, 0);
    v16h bf = load_b_frag(&ks_[nt * 16][0], Cc, lane, 0);
    acc = __builtin_amdgcn_wmma_f32_16x16x32_f16(false, af, false, bf,
                                                 (short)0, acc, false, false);
    int col = nt * 16 + (lane & 15);
    int rb  = mt * 16 + ((lane >> 4) * 8);
#pragma unroll
    for (int r = 0; r < 8; ++r) sc[rb + r][col] = acc[r] * scale;
  }
  __syncthreads();

  if (tid < 80) {                              // f32 softmax per row
    if (tid < Sc) {
      float mx = -3.0e38f;
      for (int c = 0; c < Sc; ++c) mx = fmaxf(mx, sc[tid][c]);
      float den = 0.f;
      for (int c = 0; c < Sc; ++c) den += expf(sc[tid][c] - mx);
      float inv = 1.f / den;
      for (int c = 0; c < 96; ++c)
        pl[tid][c] = (c < Sc) ? (half_t)(expf(sc[tid][c] - mx) * inv)
                              : (half_t)0.f;
    } else {
      for (int c = 0; c < 96; ++c) pl[tid][c] = (half_t)0.f;
    }
  }
  __syncthreads();

  for (int t = wv; t < 10; t += 8) {           // O = P @ V  (80x32, K=96)
    int mt = t >> 1, nt = t & 1;
    v8f acc;
#pragma unroll
    for (int r = 0; r < 8; ++r) acc[r] = 0.f;
#pragma unroll
    for (int kc = 0; kc < 96; kc += 32) {
      v16h af = load_a_frag(&pl[mt * 16][0], 96, lane, kc);
      v16h bf = load_b_frag(&vt[nt * 16][0], 96, lane, kc);
      acc = __builtin_amdgcn_wmma_f32_16x16x32_f16(false, af, false, bf,
                                                   (short)0, acc, false, false);
    }
    int col = nt * 16 + (lane & 15);
    int rb  = mt * 16 + ((lane >> 4) * 8);
#pragma unroll
    for (int r = 0; r < 8; ++r) {
      int row = rb + r;
      if (row < Sc)
        o[((size_t)b * Sc + row) * Dc + h * Cc + col] = (half_t)acc[r];
    }
  }
}

// ---- kernel 5: K small heads on the last token (f32, tiny) -----------------
__global__ void __launch_bounds__(64) head_kernel(
    const float* __restrict__ t32, const float* __restrict__ h1_w,
    const float* __restrict__ h1_b, const float* __restrict__ h2_w,
    const float* __restrict__ h2_b, float* __restrict__ out) {
  __shared__ float h1s[Dc / 2];
  const int b = blockIdx.x, k = blockIdx.y, o = threadIdx.x;   // o < 64
  const float* xr = t32 + ((size_t)b * Sc + (Sc - 1)) * Dc;
  float acc = h1_b[k * (Dc / 2) + o];
  for (int d = 0; d < Dc; ++d)
    acc += xr[d] * h1_w[((size_t)k * (Dc / 2) + o) * Dc + d];
  h1s[o] = acc > 0.f ? acc : 0.f;
  __syncthreads();
  if (o < NSc) {
    float a2 = h2_b[k * NSc + o];
    for (int j = 0; j < Dc / 2; ++j)
      a2 += h1s[j] * h2_w[((size_t)k * NSc + o) * (Dc / 2) + j];
    out[((size_t)k * Bc + b) * NSc + o] = a2;
  }
}

// ---- host orchestration ----------------------------------------------------
extern "C" void kernel_launch(void* const* d_in, const int* in_sizes, int n_in,
                              void* d_out, int out_size, void* d_ws,
                              size_t ws_size, hipStream_t stream) {
  const float* x        = (const float*)d_in[0];
  // d_in[1] = edge_index: dense K5 + self loops, encoded statically in kernel
  const float* gat_w    = (const float*)d_in[2];
  const float* gat_as   = (const float*)d_in[3];
  const float* gat_ad   = (const float*)d_in[4];
  const float* gat_bias = (const float*)d_in[5];
  const float* gat_ln_g = (const float*)d_in[6];
  const float* gat_ln_b = (const float*)d_in[7];
  const float* qkv_w    = (const float*)d_in[8];
  const float* qkv_b    = (const float*)d_in[9];
  const float* out_w    = (const float*)d_in[10];
  const float* out_b    = (const float*)d_in[11];
  const float* ln1_g    = (const float*)d_in[12];
  const float* ln1_b    = (const float*)d_in[13];
  const float* ff1_w    = (const float*)d_in[14];
  const float* ff1_b    = (const float*)d_in[15];
  const float* ff2_w    = (const float*)d_in[16];
  const float* ff2_b    = (const float*)d_in[17];
  const float* ln2_g    = (const float*)d_in[18];
  const float* ln2_b    = (const float*)d_in[19];
  const float* h1_w     = (const float*)d_in[20];
  const float* h1_b     = (const float*)d_in[21];
  const float* h2_w     = (const float*)d_in[22];
  const float* h2_b     = (const float*)d_in[23];

  char* ws = (char*)d_ws;
  size_t off = 0;
  float*  t32   = (float*)(ws + off);  off += (size_t)Gc * Dc * 4;
  half_t* t16   = (half_t*)(ws + off); off += (size_t)Gc * Dc * 2;
  half_t* qkv16 = (half_t*)(ws + off); off += (size_t)Gc * 3 * Dc * 2;
  half_t* ao16  = (half_t*)(ws + off); off += (size_t)Gc * Dc * 2;
  half_t* ff16  = (half_t*)(ws + off); off += (size_t)Gc * FFc * 2;
  half_t* wq16  = (half_t*)(ws + off); off += (size_t)Lc * 3 * Dc * Dc * 2;
  half_t* wo16  = (half_t*)(ws + off); off += (size_t)Lc * Dc * Dc * 2;
  half_t* wf1   = (half_t*)(ws + off); off += (size_t)Lc * FFc * Dc * 2;
  half_t* wf2   = (half_t*)(ws + off); off += (size_t)Lc * Dc * FFc * 2;

  // weight mirrors (once per call; weights are L2-resident afterwards)
  {
    int n;
    n = Lc * 3 * Dc * Dc;
    f32_to_f16_kernel<<<(n + 255) / 256, 256, 0, stream>>>(qkv_w, wq16, n);
    n = Lc * Dc * Dc;
    f32_to_f16_kernel<<<(n + 255) / 256, 256, 0, stream>>>(out_w, wo16, n);
    n = Lc * FFc * Dc;
    f32_to_f16_kernel<<<(n + 255) / 256, 256, 0, stream>>>(ff1_w, wf1, n);
    n = Lc * Dc * FFc;
    f32_to_f16_kernel<<<(n + 255) / 256, 256, 0, stream>>>(ff2_w, wf2, n);
  }

  gat_pe_kernel<<<Gc / 8, 256, 0, stream>>>(x, gat_w, gat_as, gat_ad, gat_bias,
                                            gat_ln_g, gat_ln_b, t32, t16);

  for (int l = 0; l < Lc; ++l) {
    gemm_bias_f16_kernel<<<dim3(Gc / 16, 3), 256, 0, stream>>>(
        t16, wq16 + (size_t)l * 3 * Dc * Dc, qkv_b + l * 3 * Dc, qkv16,
        Gc, Dc, 3 * Dc, 0);
    attn_kernel<<<dim3(Hc, Bc), 256, 0, stream>>>(qkv16, ao16);
    gemm_res_ln_kernel<<<Gc / 16, 256, 0, stream>>>(
        ao16, wo16 + (size_t)l * Dc * Dc, out_b + l * Dc,
        ln1_g + l * Dc, ln1_b + l * Dc, t32, t16, Dc);
    gemm_bias_f16_kernel<<<dim3(Gc / 16, 2), 256, 0, stream>>>(
        t16, wf1 + (size_t)l * FFc * Dc, ff1_b + l * FFc, ff16,
        Gc, Dc, FFc, 1);
    gemm_res_ln_kernel<<<Gc / 16, 256, 0, stream>>>(
        ff16, wf2 + (size_t)l * Dc * FFc, ff2_b + l * Dc,
        ln2_g + l * Dc, ln2_b + l * Dc, t32, t16, FFc);
  }

  head_kernel<<<dim3(Bc, Kc), 64, 0, stream>>>(t32, h1_w, h1_b, h2_w, h2_b,
                                               (float*)d_out);
}
